// RecurrentDecoder_58806692217074
// MI455X (gfx1250) — matrix operand order repaired and checked by reading, not compiled, not run on hardware
//
#include <hip/hip_runtime.h>

// ---------------------------------------------------------------------------
// CDNA5 (gfx1250) GRU decoder: bf16 WMMA pipeline, LDS-staged logits GEMM,
// persistent-kernel recurrence with device-wide atomic barrier.
// ---------------------------------------------------------------------------

typedef __attribute__((ext_vector_type(16))) __bf16 v16bf;
typedef __attribute__((ext_vector_type(8)))  __bf16 v8bf;
typedef __attribute__((ext_vector_type(8)))  float  v8f;

#define B_   16
#define TT_  128
#define TS_  127          // teacher-forced steps
#define E_   512
#define U_   1024
#define V_   32000
#define M_   (B_ * TS_)   // 2032 GEMM rows
#define NBLK_GRU 64       // persistent blocks for recurrence (one 16-wide U tile each)

#if defined(__AMDGCN__) && __has_builtin(__builtin_amdgcn_global_load_async_to_lds_b128) && \
    __has_builtin(__builtin_amdgcn_s_wait_asynccnt)
#define HAVE_ASYNC_LDS 1
typedef __attribute__((ext_vector_type(4))) int v4i;
typedef v4i __attribute__((address_space(1)))* gptr128;   // global-side operand
typedef v4i __attribute__((address_space(3)))* lptr128;   // LDS-side operand
#else
#define HAVE_ASYNC_LDS 0
#endif

__device__ __forceinline__ __bf16 f2bf(float f) {
  union { float f; unsigned u; } v; v.f = f;
  unsigned u = v.u + 0x7fffu + ((v.u >> 16) & 1u);   // round-to-nearest-even
  unsigned short s = (unsigned short)(u >> 16);
  __bf16 r; __builtin_memcpy(&r, &s, 2);
  return r;
}

// A operand (16x32 bf16, row-major source): lane = M row; lanes 0-15 take
// K {kk+0..7, kk+16..23}, lanes 16-31 take K {kk+8..15, kk+24..31}.
__device__ __forceinline__ v16bf load_a(const __bf16* __restrict__ A, int lda,
                                        int m_base, int kk, int lane) {
  int m  = m_base + (lane & 15);
  int kh = (lane >> 4) << 3;
  const __bf16* row = A + (size_t)m * lda + kk + kh;
  v8bf lo = *(const v8bf*)(row);
  v8bf hi = *(const v8bf*)(row + 16);
  v16bf a;
#pragma unroll
  for (int i = 0; i < 8; ++i) { a[i] = lo[i]; a[8 + i] = hi[i]; }
  return a;
}

// B operand: pre-packed, each lane reads 16 contiguous bf16 (32 bytes).
// Pack layout: [nt][kt][lane 0..31][elem 0..15].
__device__ __forceinline__ v16bf load_b(const __bf16* __restrict__ Bp, int KT,
                                        int nt, int kt, int lane) {
  return *(const v16bf*)(Bp + ((((size_t)nt * KT + kt) * 32 + lane) << 4));
}

__device__ __forceinline__ v8f wmma_bf16(v16bf a, v16bf b, v8f c) {
  return __builtin_amdgcn_wmma_f32_16x16x32_bf16(false, a, false, b,
                                                 (short)0, c, false, false);
}

// --- pack f32 [K x N] row-major into WMMA B layout, bf16 --------------------
__global__ void k_pack_b(const float* __restrict__ src, __bf16* __restrict__ dst,
                         int K, int N) {
  long long idx = (long long)blockIdx.x * blockDim.x + threadIdx.x;
  long long total = (long long)K * N;
  if (idx >= total) return;
  int e = (int)(idx & 15);
  int l = (int)((idx >> 4) & 31);
  long long rest = idx >> 9;
  int KT = K >> 5;
  int kt = (int)(rest % KT);
  int nt = (int)(rest / KT);
  int k = (kt << 5) + ((l >> 4) << 3) + (e & 7) + ((e >> 3) << 4);
  int n = (nt << 4) + (l & 15);
  dst[idx] = f2bf(src[(size_t)k * N + n]);
}

// --- gather teacher tokens -> bf16 A matrix [M_ x E_] -----------------------
__global__ void k_gather_x(const int* __restrict__ tokens, const float* __restrict__ emb,
                           __bf16* __restrict__ xb) {
  int idx = blockIdx.x * blockDim.x + threadIdx.x;
  if (idx >= M_ * E_) return;
  int m = idx >> 9;
  int e = idx & 511;
  int b = m / TS_, t = m % TS_;
  int tok = tokens[b * TT_ + t];
  xb[idx] = f2bf(emb[(size_t)tok * E_ + e]);
}

// --- h0 = latent @ W_dense + b_dense (tiny) ---------------------------------
__global__ void k_h0(const float* __restrict__ latent, const float* __restrict__ Wd,
                     const float* __restrict__ bd, float* __restrict__ h,
                     __bf16* __restrict__ hbf0) {
  int idx = blockIdx.x * blockDim.x + threadIdx.x;
  if (idx >= B_ * U_) return;
  int b = idx >> 10, u = idx & 1023;
  float acc = bd[u];
  for (int k = 0; k < 512; ++k) acc += latent[b * 512 + k] * Wd[(size_t)k * U_ + u];
  h[idx] = acc;
  hbf0[idx] = f2bf(acc);
}

// --- init the device-wide barrier counter -----------------------------------
__global__ void k_init(unsigned* __restrict__ bar) { *bar = 0u; }

// --- xp = x @ W_x + b_x for all timesteps, f32 out [M_ x 3U] ----------------
__global__ void k_xp(const __bf16* __restrict__ xb, const __bf16* __restrict__ Wxp,
                     const float* __restrict__ bx, float* __restrict__ xp) {
  const int NT = (3 * U_) / 16, MT = M_ / 16, KT = E_ / 32;
  int lane = threadIdx.x & 31;
  int wave = blockIdx.x * (blockDim.x >> 5) + (threadIdx.x >> 5);
  int nwaves = gridDim.x * (blockDim.x >> 5);
  for (int tile = wave; tile < MT * NT; tile += nwaves) {
    int mt = tile / NT, nt = tile % NT;
    v8f c = {};
    for (int kt = 0; kt < KT; ++kt) {
      v16bf a = load_a(xb, E_, mt * 16, kt * 32, lane);
      v16bf b = load_b(Wxp, KT, nt, kt, lane);
      c = wmma_bf16(a, b, c);
    }
    int n = nt * 16 + (lane & 15);
    float bias = bx[n];
    int mrow = mt * 16 + ((lane >> 4) << 3);
#pragma unroll
    for (int r = 0; r < 8; ++r)
      xp[(size_t)(mrow + r) * (3 * U_) + n] = c[r] + bias;
  }
}

// --- persistent GRU recurrence: 64 blocks x 1 wave, 127 steps in one launch -
__global__ void k_gru_all(const __bf16* __restrict__ Whp, const float* __restrict__ bh,
                          const float* __restrict__ xp, const int* __restrict__ tokens,
                          float* __restrict__ h, __bf16* __restrict__ hbf0,
                          __bf16* __restrict__ hbf1, __bf16* __restrict__ grub,
                          unsigned* __restrict__ bar) {
  const int KT = U_ / 32;
  int lane = threadIdx.x & 31;
  int jt = blockIdx.x;                    // 64 tiles over U_
  int j = jt * 16 + (lane & 15);
  float biz = bh[j], bir = bh[U_ + j], bih = bh[2 * U_ + j];

  for (int t = 0; t < TS_; ++t) {
    const __bf16* hin  = (t & 1) ? hbf1 : hbf0;
    __bf16*       hout = (t & 1) ? hbf0 : hbf1;
    v8f cz = {}, cr = {}, ch = {};
    for (int kt = 0; kt < KT; ++kt) {
      v16bf a  = load_a(hin, U_, 0, kt * 32, lane);
      v16bf bz = load_b(Whp, KT, jt,       kt, lane);
      v16bf br = load_b(Whp, KT, jt + 64,  kt, lane);
      v16bf bq = load_b(Whp, KT, jt + 128, kt, lane);
      cz = wmma_bf16(a, bz, cz);
      cr = wmma_bf16(a, br, cr);
      ch = wmma_bf16(a, bq, ch);
    }
#pragma unroll
    for (int r = 0; r < 8; ++r) {
      int b = r + ((lane >> 4) << 3);
      size_t mrow = (size_t)b * TS_ + t;
      const float* xrow = xp + mrow * (3 * U_);
      float xz = xrow[j], xr = xrow[U_ + j], xh = xrow[2 * U_ + j];
      float hz = cz[r] + biz, hr = cr[r] + bir, hh = ch[r] + bih;
      float z  = 1.f / (1.f + __expf(-(xz + hz)));
      float rg = 1.f / (1.f + __expf(-(xr + hr)));
      float hc = tanhf(xh + rg * hh);
      float hp = h[b * U_ + j];
      float hn = z * hp + (1.f - z) * hc;
      float ho = (tokens[b * TT_ + t] != 0) ? hn : hp;
      h[b * U_ + j] = ho;
      __bf16 hb = f2bf(ho);
      hout[b * U_ + j] = hb;
      grub[mrow * U_ + j] = hb;
    }
    // device-wide split barrier: monotonic ticket counter in L2
    __threadfence();
    if (lane == 0) {
      __hip_atomic_fetch_add(bar, 1u, __ATOMIC_RELEASE, __HIP_MEMORY_SCOPE_AGENT);
      unsigned tgt = (unsigned)(t + 1) * (unsigned)NBLK_GRU;
      while (__hip_atomic_load(bar, __ATOMIC_ACQUIRE, __HIP_MEMORY_SCOPE_AGENT) < tgt)
        __builtin_amdgcn_s_sleep(1);
    }
    __syncthreads();
    __threadfence();
  }
}

// --- logits = gru_out @ W_out + b_out (133 GFLOP, dominant) -----------------
// Block = 8 waves; A block (16x1024) staged in LDS (async copy when available),
// each wave owns a 16x64 output strip: 4 independent WMMA chains, pipelined.
#define APAD 1032   // padded LDS row stride (elements): 16B aligned, conflict-free
__global__ void k_logits(const __bf16* __restrict__ Ab, const __bf16* __restrict__ Wop,
                         const float* __restrict__ bo, float* __restrict__ out) {
  const int NT = V_ / 16, KT = U_ / 32;   // 2000, 32
  __shared__ __bf16 As[16 * APAD];        // 33 KB
  int lane = threadIdx.x & 31;
  int widx = threadIdx.x >> 5;            // 0..7
  int mt = blockIdx.x % (M_ / 16);        // all mt of one N-slab concurrent -> L2 reuse
  int bn = blockIdx.x / (M_ / 16);

  // stage A block (16 rows x 1024 cols) into padded LDS
  const __bf16* Ablk = Ab + (size_t)mt * 16 * U_;
  for (int f = threadIdx.x * 8; f < 16 * U_; f += 256 * 8) {
    int m = f >> 10, col = f & 1023;
#if HAVE_ASYNC_LDS
    __builtin_amdgcn_global_load_async_to_lds_b128(
        (gptr128)(Ablk + f), (lptr128)(As + m * APAD + col), 0, 0);
#else
    *(v8bf*)(As + m * APAD + col) = *(const v8bf*)(Ablk + f);
#endif
  }
#if HAVE_ASYNC_LDS
  __builtin_amdgcn_s_wait_asynccnt(0);
#endif
  __syncthreads();

  int nt0 = bn * 32 + widx * 4;
  if (nt0 >= NT) return;                  // wave-uniform (after barrier)
  int nt1 = (nt0 + 1 < NT) ? nt0 + 1 : NT - 1;
  int nt2 = (nt0 + 2 < NT) ? nt0 + 2 : NT - 1;
  int nt3 = (nt0 + 3 < NT) ? nt0 + 3 : NT - 1;
  const __bf16* bp0 = Wop + ((size_t)nt0 * KT << 9);
  const __bf16* bp1 = Wop + ((size_t)nt1 * KT << 9);
  const __bf16* bp2 = Wop + ((size_t)nt2 * KT << 9);
  const __bf16* bp3 = Wop + ((size_t)nt3 * KT << 9);

  int abase = (lane & 15) * APAD + ((lane >> 4) << 3);
  auto lda = [&](int kt) {
    v8bf lo = *(const v8bf*)(As + abase + kt * 32);
    v8bf hi = *(const v8bf*)(As + abase + kt * 32 + 16);
    v16bf a;
#pragma unroll
    for (int i = 0; i < 8; ++i) { a[i] = lo[i]; a[8 + i] = hi[i]; }
    return a;
  };
  auto ldb = [&](const __bf16* bp, int kt) {
    return *(const v16bf*)(bp + (((size_t)kt * 32 + lane) << 4));
  };

  v8f c0 = {}, c1 = {}, c2 = {}, c3 = {};
  v16bf a  = lda(0);
  v16bf b0 = ldb(bp0, 0), b1 = ldb(bp1, 0), b2 = ldb(bp2, 0), b3 = ldb(bp3, 0);
  for (int kt = 0; kt < KT - 1; ++kt) {
    if (kt + 3 < KT) {                    // stream ahead -> global_prefetch_b8
      __builtin_prefetch((const void*)(bp0 + (((size_t)(kt + 3) * 32 + lane) << 4)), 0, 1);
      __builtin_prefetch((const void*)(bp1 + (((size_t)(kt + 3) * 32 + lane) << 4)), 0, 1);
      __builtin_prefetch((const void*)(bp2 + (((size_t)(kt + 3) * 32 + lane) << 4)), 0, 1);
      __builtin_prefetch((const void*)(bp3 + (((size_t)(kt + 3) * 32 + lane) << 4)), 0, 1);
    }
    v16bf an  = lda(kt + 1);
    v16bf b0n = ldb(bp0, kt + 1), b1n = ldb(bp1, kt + 1);
    v16bf b2n = ldb(bp2, kt + 1), b3n = ldb(bp3, kt + 1);
    c0 = wmma_bf16(a, b0, c0);
    c1 = wmma_bf16(a, b1, c1);
    c2 = wmma_bf16(a, b2, c2);
    c3 = wmma_bf16(a, b3, c3);
    a = an; b0 = b0n; b1 = b1n; b2 = b2n; b3 = b3n;
  }
  c0 = wmma_bf16(a, b0, c0);
  c1 = wmma_bf16(a, b1, c1);
  c2 = wmma_bf16(a, b2, c2);
  c3 = wmma_bf16(a, b3, c3);

  int nl = lane & 15;
  int mrow = mt * 16 + ((lane >> 4) << 3);
  v8f cc[4] = {c0, c1, c2, c3};
#pragma unroll
  for (int i = 0; i < 4; ++i) {
    int nt = nt0 + i;
    if (nt >= NT) break;                  // wave-uniform
    int n = nt * 16 + nl;
    float bias = bo[n];
#pragma unroll
    for (int r = 0; r < 8; ++r)
      out[(size_t)(mrow + r) * V_ + n] = cc[i][r] + bias;
  }
}

// ---------------------------------------------------------------------------
extern "C" void kernel_launch(void* const* d_in, const int* in_sizes, int n_in,
                              void* d_out, int out_size, void* d_ws, size_t ws_size,
                              hipStream_t stream) {
  const float* latent = (const float*)d_in[0];
  const int*   tokens = (const int*)d_in[1];
  const float* emb    = (const float*)d_in[2];
  const float* Wd     = (const float*)d_in[3];
  const float* bd     = (const float*)d_in[4];
  const float* Wx     = (const float*)d_in[5];
  const float* Wh     = (const float*)d_in[6];
  const float* bx     = (const float*)d_in[7];
  const float* bhh    = (const float*)d_in[8];
  const float* Wo     = (const float*)d_in[9];
  const float* bo     = (const float*)d_in[10];
  float* out = (float*)d_out;

  char* ws = (char*)d_ws;
  size_t off = 0;
  auto carve = [&](size_t bytes) -> char* {
    char* p = ws + off;
    off += (bytes + 255) & ~(size_t)255;
    return p;
  };
  __bf16*   Wxp  = (__bf16*)carve((size_t)E_ * 3 * U_ * 2);   // 3.1 MB
  __bf16*   Whp  = (__bf16*)carve((size_t)U_ * 3 * U_ * 2);   // 6.3 MB
  __bf16*   Wop  = (__bf16*)carve((size_t)U_ * V_ * 2);       // 65.5 MB
  __bf16*   xb   = (__bf16*)carve((size_t)M_ * E_ * 2);       // 2.1 MB
  __bf16*   grub = (__bf16*)carve((size_t)M_ * U_ * 2);       // 4.2 MB
  __bf16*   hbf0 = (__bf16*)carve((size_t)B_ * U_ * 2);
  __bf16*   hbf1 = (__bf16*)carve((size_t)B_ * U_ * 2);
  float*    h    = (float*)carve((size_t)B_ * U_ * 4);
  float*    xp   = (float*)carve((size_t)M_ * 3 * U_ * 4);    // 25 MB
  unsigned* bar  = (unsigned*)carve(256);

  // 1) pack weights to bf16 WMMA-B layout; init barrier
  k_init<<<1, 1, 0, stream>>>(bar);
  {
    long long tot = (long long)E_ * 3 * U_;
    k_pack_b<<<(unsigned)((tot + 255) / 256), 256, 0, stream>>>(Wx, Wxp, E_, 3 * U_);
  }
  {
    long long tot = (long long)U_ * 3 * U_;
    k_pack_b<<<(unsigned)((tot + 255) / 256), 256, 0, stream>>>(Wh, Whp, U_, 3 * U_);
  }
  {
    long long tot = (long long)U_ * V_;
    k_pack_b<<<(unsigned)((tot + 255) / 256), 256, 0, stream>>>(Wo, Wop, U_, V_);
  }

  // 2) gather embeddings, initial hidden state
  k_gather_x<<<(M_ * E_ + 255) / 256, 256, 0, stream>>>(tokens, emb, xb);
  k_h0<<<(B_ * U_ + 255) / 256, 256, 0, stream>>>(latent, Wd, bd, h, hbf0);

  // 3) xp for all timesteps (WMMA)
  k_xp<<<1024, 256, 0, stream>>>(xb, Wxp, bx, xp);

  // 4) whole recurrence in ONE persistent kernel (64 co-resident blocks)
  k_gru_all<<<NBLK_GRU, 32, 0, stream>>>(Whp, bhh, xp, tokens, h, hbf0, hbf1,
                                         grub, bar);

  // 5) logits GEMM: 127 mt x 63 n-slabs, LDS-staged A, 16x64 per wave
  k_logits<<<(M_ / 16) * 63, 256, 0, stream>>>(grub, Wop, bo, out);
}
